// MultiHeadAttention_4595615006793
// MI455X (gfx1250) — matrix operand order repaired and checked by reading, not compiled, not run on hardware
//
#include <hip/hip_runtime.h>
#include <hip/hip_bf16.h>
#include <math.h>

// Problem constants (from reference)
#define CC 1024        // N_STATE
#define TT 4096        // T
#define BB 4           // B
#define BT (BB*TT)     // tokens = 16384
#define HH 16          // heads
#define DH 64          // head dim
#define NCH 32         // scan chunks over T
#define TC (TT/NCH)    // 128 steps per chunk

typedef __attribute__((ext_vector_type(16))) _Float16 v16h;
typedef __attribute__((ext_vector_type(8)))  _Float16 v8h;
typedef __attribute__((ext_vector_type(8)))  float    v8f;

// ---------------------------------------------------------------------------
// A-fragment: 16x32 f16, rows = tokens (M = lane&15), K contiguous (matches B):
//   lanes 0-15 : halves 0..15 = K+0..15
//   lanes 16-31: halves 0..15 = K+16..31
// Two contiguous 16B loads per lane from row-major (BT x CC) f16.
__device__ __forceinline__ v16h load_a_frag(const _Float16* __restrict__ Xh,
                                            int row, int kbase, int lane) {
    int hi = (lane >> 4) * 16;
    const _Float16* p = Xh + (size_t)row * CC + kbase + hi;
    v8h a0 = *(const v8h*)(p);
    v8h a1 = *(const v8h*)(p + 8);
    v16h a;
#pragma unroll
    for (int i = 0; i < 8; ++i) { a[i] = a0[i]; a[i + 8] = a1[i]; }
    return a;
}

// B-fragment: 32x16 f16, B[k][n] = W[n][k] (W row-major O x C).
//   lane L: column N = L&15, K-offset = (L>=16)?16:0, 16 contiguous halves.
// One contiguous 32B load per lane.
__device__ __forceinline__ v16h load_b_frag(const _Float16* __restrict__ Wh,
                                            int nrow, int kbase, int lane) {
    const _Float16* p = Wh + (size_t)nrow * CC + kbase + ((lane >> 4) << 4);
    return *(const v16h*)p;
}

__device__ __forceinline__ float phi_act(float y) {
    // silu(y) + 1
    return y / (1.0f + expf(-y)) + 1.0f;
}

// ---------------------------------------------------------------------------
// Kernel 1: convert Wq|Wk|Wv|Wo (each 1M f32) -> f16, concatenated.
__global__ __launch_bounds__(256) void conv_w_kernel(
    const float* __restrict__ Wq, const float* __restrict__ Wk,
    const float* __restrict__ Wv, const float* __restrict__ Wo,
    _Float16* __restrict__ Wh) {
    size_t i = (size_t)blockIdx.x * 256 + threadIdx.x;   // 0 .. 4M-1
    int which = (int)(i >> 20);
    size_t off = i & ((1u << 20) - 1);
    const float* src = (which == 0) ? Wq : (which == 1) ? Wk : (which == 2) ? Wv : Wo;
    Wh[i] = (_Float16)src[off];
}

// ---------------------------------------------------------------------------
// Kernel 2: LDS-tiled transpose+convert x (B,C,T) f32 -> Xh (B*T, C) f16.
__global__ __launch_bounds__(256) void xpose_kernel(
    const float* __restrict__ x, _Float16* __restrict__ Xh) {
    __shared__ float tile[32][33];
    int b  = blockIdx.z;
    int t0 = blockIdx.x * 32;
    int c0 = blockIdx.y * 32;
    int tx = threadIdx.x, ty = threadIdx.y;  // block (32,8)
#pragma unroll
    for (int k = 0; k < 32; k += 8)
        tile[ty + k][tx] = x[((size_t)(b * CC + c0 + ty + k)) * TT + t0 + tx];
    __syncthreads();
#pragma unroll
    for (int k = 0; k < 32; k += 8)
        Xh[((size_t)(b * TT + t0 + ty + k)) * CC + c0 + tx] = (_Float16)tile[tx][ty + k];
}

// ---------------------------------------------------------------------------
// Kernel 3: fused QKV GEMM + bias + phi.
// Register-blocked: each wave computes a 32(M) x 64(N) tile:
//   2 A-frags x 4 B-frags -> 8 WMMAs per 32-deep K-step (1.5 b128 loads/WMMA).
// grid = (BT/32, 3*1024/(8*64)); block = 256 (8 waves).
__global__ __launch_bounds__(256) void qkv_gemm_kernel(
    const _Float16* __restrict__ Xh,
    const _Float16* __restrict__ Whq, const _Float16* __restrict__ Whk,
    const _Float16* __restrict__ Whv,
    const float* __restrict__ bq, const float* __restrict__ bv,
    float* __restrict__ Qb, float* __restrict__ Kb, float* __restrict__ Vb) {
    int lane = threadIdx.x & 31;
    int wave = threadIdx.x >> 5;
    int mbase = blockIdx.x * 32;
    int ncat = (blockIdx.y * 8 + wave) * 64;   // 0..3008, never crosses a 1024 boundary
    int sel  = ncat >> 10;                      // 0=q, 1=k, 2=v
    int nloc = ncat & 1023;

    const _Float16* Wh = (sel == 0) ? Whq : (sel == 1) ? Whk : Whv;
    const float* bias  = (sel == 0) ? bq : (sel == 2) ? bv : nullptr;
    float* Out         = (sel == 0) ? Qb : (sel == 1) ? Kb : Vb;

    v8f acc[2][4];
#pragma unroll
    for (int i = 0; i < 2; ++i)
#pragma unroll
        for (int j = 0; j < 4; ++j) acc[i][j] = (v8f){};

    int arow = mbase + (lane & 15);
    int nrow = nloc + (lane & 15);

    for (int k = 0; k < CC; k += 32) {
        // prefetch next K-step (lowers to global_prefetch_b8)
        if (k + 32 < CC) {
            __builtin_prefetch(Xh + (size_t)arow * CC + k + 32, 0, 1);
            __builtin_prefetch(Wh + (size_t)nrow * CC + k + 32, 0, 1);
        }
        v16h a0 = load_a_frag(Xh, arow, k, lane);
        v16h a1 = load_a_frag(Xh, arow + 16, k, lane);
#pragma unroll
        for (int j = 0; j < 4; ++j) {
            v16h bf = load_b_frag(Wh, nrow + j * 16, k, lane);
            acc[0][j] = __builtin_amdgcn_wmma_f32_16x16x32_f16(
                false, a0, false, bf, (short)0, acc[0][j], false, false);
            acc[1][j] = __builtin_amdgcn_wmma_f32_16x16x32_f16(
                false, a1, false, bf, (short)0, acc[1][j], false, false);
        }
    }

#pragma unroll
    for (int i = 0; i < 2; ++i) {
        int rbase = mbase + i * 16 + ((lane >> 4) ? 8 : 0);
#pragma unroll
        for (int j = 0; j < 4; ++j) {
            int col   = nloc + j * 16 + (lane & 15);
            float bvl = bias ? bias[col] : 0.0f;
#pragma unroll
            for (int r = 0; r < 8; ++r) {
                float y = acc[i][j][r] + bvl;
                if (sel < 2) y = phi_act(y);
                Out[(size_t)(rbase + r) * CC + col] = y;
            }
        }
    }
}

// ---------------------------------------------------------------------------
// Kernel 4: per-chunk partial sums of phi_k and phi_k*v.
// One wave32 per (b,h,chunk); each lane handles channels d=lane and d=lane+32.
__global__ __launch_bounds__(256) void scan_part_kernel(
    const float* __restrict__ Kb, const float* __restrict__ Vb,
    float* __restrict__ ksumC, float* __restrict__ kvsumC) {
    int lane = threadIdx.x & 31;
    int wave = threadIdx.x >> 5;
    int unit = blockIdx.x * 8 + wave;    // 0 .. B*H*NCH-1 = 2047
    int chunk = unit & (NCH - 1);
    int bh = unit >> 5;                  // NCH = 32
    int b = bh >> 4, h = bh & 15;
    int cbase = h * DH;

    float ks0 = 0.f, ks1 = 0.f, kv0 = 0.f, kv1 = 0.f;
    for (int t = chunk * TC; t < (chunk + 1) * TC; ++t) {
        size_t roff = ((size_t)(b * TT + t)) * CC + cbase;
        float pk0 = Kb[roff + lane],      pk1 = Kb[roff + lane + 32];
        float v0  = Vb[roff + lane],      v1  = Vb[roff + lane + 32];
        ks0 += pk0; ks1 += pk1; kv0 += pk0 * v0; kv1 += pk1 * v1;
    }
    size_t so = ((size_t)bh * NCH + chunk) * DH;
    ksumC[so + lane] = ks0;       ksumC[so + lane + 32] = ks1;
    kvsumC[so + lane] = kv0;      kvsumC[so + lane + 32] = kv1;
}

// Kernel 5: exclusive prefix over the 32 chunks, per (b,h,d) channel.
__global__ __launch_bounds__(256) void scan_prefix_kernel(
    float* __restrict__ ksumC, float* __restrict__ kvsumC) {
    int tid = blockIdx.x * 256 + threadIdx.x;  // 0..4095
    int bh = tid >> 6, d = tid & 63;
    float r0 = 0.f, r1 = 0.f;
    for (int c = 0; c < NCH; ++c) {
        size_t idx = ((size_t)bh * NCH + c) * DH + d;
        float t0 = ksumC[idx];  ksumC[idx]  = r0; r0 += t0;
        float t1 = kvsumC[idx]; kvsumC[idx] = r1; r1 += t1;
    }
}

// Kernel 6: final scan — running cumsum + wave-reduced denominator,
// writes attention output (pre-Wo, scaled by 1/sqrt(Dh)) as f16 token-major.
__global__ __launch_bounds__(256) void scan_final_kernel(
    const float* __restrict__ Qb, const float* __restrict__ Kb,
    const float* __restrict__ Vb,
    const float* __restrict__ ksumC, const float* __restrict__ kvsumC,
    _Float16* __restrict__ Ah) {
    int lane = threadIdx.x & 31;
    int wave = threadIdx.x >> 5;
    int unit = blockIdx.x * 8 + wave;
    int chunk = unit & (NCH - 1);
    int bh = unit >> 5;
    int b = bh >> 4, h = bh & 15;
    int cbase = h * DH;

    size_t so = ((size_t)bh * NCH + chunk) * DH;
    float ks0 = ksumC[so + lane],  ks1 = ksumC[so + lane + 32];
    float kv0 = kvsumC[so + lane], kv1 = kvsumC[so + lane + 32];

    for (int t = chunk * TC; t < (chunk + 1) * TC; ++t) {
        size_t roff = ((size_t)(b * TT + t)) * CC + cbase;
        float pk0 = Kb[roff + lane],      pk1 = Kb[roff + lane + 32];
        float v0  = Vb[roff + lane],      v1  = Vb[roff + lane + 32];
        ks0 += pk0; kv0 += pk0 * v0;
        ks1 += pk1; kv1 += pk1 * v1;
        float pq0 = Qb[roff + lane], pq1 = Qb[roff + lane + 32];
        float dpart = pq0 * ks0 + pq1 * ks1;
#pragma unroll
        for (int off = 16; off > 0; off >>= 1)
            dpart += __shfl_down(dpart, off, 32);
        float den = __shfl(dpart, 0, 32) + 1e-6f;
        float inv = 0.125f / den;               // includes 1/sqrt(Dh)
        Ah[roff + lane]      = (_Float16)(pq0 * kv0 * inv);
        Ah[roff + lane + 32] = (_Float16)(pq1 * kv1 * inv);
    }
}

// ---------------------------------------------------------------------------
// Kernel 7: output GEMM (Wo + bo), register-blocked 32x64 per wave,
// scatter D back into the reference (B, C, T) layout.
__global__ __launch_bounds__(256) void out_gemm_kernel(
    const _Float16* __restrict__ Ah, const _Float16* __restrict__ Who,
    const float* __restrict__ bo, float* __restrict__ out) {
    int lane = threadIdx.x & 31;
    int wave = threadIdx.x >> 5;
    int mbase = blockIdx.x * 32;
    int nbase = (blockIdx.y * 8 + wave) * 64;   // 0..960

    v8f acc[2][4];
#pragma unroll
    for (int i = 0; i < 2; ++i)
#pragma unroll
        for (int j = 0; j < 4; ++j) acc[i][j] = (v8f){};

    int arow = mbase + (lane & 15);
    int nrow = nbase + (lane & 15);

    for (int k = 0; k < CC; k += 32) {
        if (k + 32 < CC) {
            __builtin_prefetch(Ah + (size_t)arow * CC + k + 32, 0, 1);
            __builtin_prefetch(Who + (size_t)nrow * CC + k + 32, 0, 1);
        }
        v16h a0 = load_a_frag(Ah, arow, k, lane);
        v16h a1 = load_a_frag(Ah, arow + 16, k, lane);
#pragma unroll
        for (int j = 0; j < 4; ++j) {
            v16h bf = load_b_frag(Who, nrow + j * 16, k, lane);
            acc[0][j] = __builtin_amdgcn_wmma_f32_16x16x32_f16(
                false, a0, false, bf, (short)0, acc[0][j], false, false);
            acc[1][j] = __builtin_amdgcn_wmma_f32_16x16x32_f16(
                false, a1, false, bf, (short)0, acc[1][j], false, false);
        }
    }

#pragma unroll
    for (int i = 0; i < 2; ++i) {
        int tbase = mbase + i * 16 + ((lane >> 4) ? 8 : 0);
#pragma unroll
        for (int j = 0; j < 4; ++j) {
            int col = nbase + j * 16 + (lane & 15);
            float bvl = bo[col];
#pragma unroll
            for (int r = 0; r < 8; ++r) {
                int token = tbase + r;
                int b = token >> 12;            // T = 4096
                int t = token & (TT - 1);
                out[((size_t)b * CC + col) * TT + t] = acc[i][j][r] + bvl;
            }
        }
    }
}

// ---------------------------------------------------------------------------
extern "C" void kernel_launch(void* const* d_in, const int* in_sizes, int n_in,
                              void* d_out, int out_size, void* d_ws, size_t ws_size,
                              hipStream_t stream) {
    const float* x  = (const float*)d_in[0];
    const float* Wq = (const float*)d_in[1];
    const float* bq = (const float*)d_in[2];
    const float* Wk = (const float*)d_in[3];
    const float* Wv = (const float*)d_in[4];
    const float* bv = (const float*)d_in[5];
    const float* Wo = (const float*)d_in[6];
    const float* bo = (const float*)d_in[7];
    float* out = (float*)d_out;

    char* ws = (char*)d_ws;
    // Workspace layout (bytes):
    //   [0,   8M)  Wh   : 4 x 1M f16 weights (q,k,v,o)
    //   [8M, 40M)  Xh   : (BT x C) f16
    //   [40M,104M) Qb   : phi_q f32
    //   [104M,168M)Kb   : phi_k f32
    //   [168M,232M)Vb   : v f32
    //   [232M,264M)Ah   : attention out f16 (BT x C)
    //   [264M,...) chunk sums (2 x 512KB)
    _Float16* Wh  = (_Float16*)(ws);
    _Float16* Xh  = (_Float16*)(ws + ((size_t)8  << 20));
    float*    Qb  = (float*)   (ws + ((size_t)40 << 20));
    float*    Kb  = (float*)   (ws + ((size_t)104 << 20));
    float*    Vb  = (float*)   (ws + ((size_t)168 << 20));
    _Float16* Ah  = (_Float16*)(ws + ((size_t)232 << 20));
    float*    ksumC  = (float*)(ws + ((size_t)264 << 20));
    float*    kvsumC = (float*)(ws + (((size_t)264 << 20) + (512u << 10)));

    const _Float16* Whq = Wh;
    const _Float16* Whk = Wh + (1u << 20);
    const _Float16* Whv = Wh + (2u << 20);
    const _Float16* Who = Wh + (3u << 20);

    conv_w_kernel<<<(4u << 20) / 256, 256, 0, stream>>>(Wq, Wk, Wv, Wo, Wh);

    xpose_kernel<<<dim3(TT / 32, CC / 32, BB), dim3(32, 8), 0, stream>>>(x, Xh);

    qkv_gemm_kernel<<<dim3(BT / 32, (3 * CC / 64) / 8), 256, 0, stream>>>(
        Xh, Whq, Whk, Whv, bq, bv, Qb, Kb, Vb);

    scan_part_kernel<<<(BB * HH * NCH) / 8, 256, 0, stream>>>(Kb, Vb, ksumC, kvsumC);
    scan_prefix_kernel<<<(BB * HH * DH) / 256, 256, 0, stream>>>(ksumC, kvsumC);
    scan_final_kernel<<<(BB * HH * NCH) / 8, 256, 0, stream>>>(Qb, Kb, Vb, ksumC, kvsumC, Ah);

    out_gemm_kernel<<<dim3(BT / 32, (CC / 64) / 8), 256, 0, stream>>>(Ah, Who, bo, out);
}
